// Transformers_44272522887235
// MI455X (gfx1250) — matrix operand order repaired and checked by reading, not compiled, not run on hardware
//
#include <hip/hip_runtime.h>
#include <math.h>
#include <cstdint>
#include <cstddef>

// Problem constants (match reference)
#define BB 2
#define SS 2048
#define EE 2048
#define HH 16
#define DD 128
#define NROWS (BB * SS)   // 4096
#define FF (4 * EE)       // 8192

typedef __attribute__((ext_vector_type(16))) __bf16        v16bf;
typedef __attribute__((ext_vector_type(8)))  float         v8f;
typedef __attribute__((ext_vector_type(4))) unsigned int   u32x4;
typedef __attribute__((ext_vector_type(8)))  int           i32x8;
typedef __attribute__((ext_vector_type(4)))  int           i32x4;

#if __has_builtin(__builtin_amdgcn_tensor_load_to_lds) && \
    __has_builtin(__builtin_amdgcn_s_wait_tensorcnt)
#define HAVE_TDM 1
#else
#define HAVE_TDM 0
#endif

__device__ __forceinline__ v8f wmma_bf16(v16bf a, v16bf b, v8f c) {
  // D = A(16x32 bf16) * B(32x16 bf16) + C(16x16 f32)
  return __builtin_amdgcn_wmma_f32_16x16x32_bf16(false, a, false, b,
                                                 (short)0, c, false, false);
}

// A-fragment: lane(m=lane%16,h=lane/16): elems 0..7 at K=h*8.., elems 8..15 at K=16+h*8..
__device__ __forceinline__ v16bf ld_frag_a(const __bf16* p) {
  union { v16bf v; u32x4 q[2]; } f;
  f.q[0] = *(const u32x4*)(p);
  f.q[1] = *(const u32x4*)(p + 16);
  return f.v;
}
// B-fragment: lane(n=lane%16,h=lane/16): elems 0..7 at K=h*8.., elems 8..15 at K=16+h*8..
__device__ __forceinline__ v16bf ld_frag_b(const __bf16* p) {
  union { v16bf v; u32x4 q[2]; } f;
  f.q[0] = *(const u32x4*)(p);
  f.q[1] = *(const u32x4*)(p + 16);
  return f.v;
}

#if HAVE_TDM
// -----------------------------------------------------------------------------
// Tensor Data Mover: DMA a 2D bf16 tile [tile1 rows x tile0 elems] from global
// (row stride = stride0 elems) into LDS at lds_off, inserting pad_amt(code)
// DWORDs of padding every pad_int(code) DWORDs (-> padded LDS row stride).
// D# per CDNA5 ISA. Issued by one wave; TENSORcnt tracks.
// 6-arg builtin form (clang-23 / therock-10.0 headers).
// -----------------------------------------------------------------------------
__device__ __forceinline__ void tdm_load_2d(unsigned lds_off, const void* g,
                                            unsigned tile0, unsigned tile1,
                                            unsigned stride0,
                                            unsigned pad_int, unsigned pad_amt) {
  const unsigned long long ga = (unsigned long long)(size_t)g;
  const unsigned tensor_d0 = stride0;       // >= tile0, keeps OOB clamp inactive
  const unsigned tensor_d1 = 1u << 20;
  u32x4 g0;
  g0[0] = 1u;                                             // count=1 (valid), user mode
  g0[1] = lds_off;                                        // lds_addr (bytes)
  g0[2] = (unsigned)(ga & 0xffffffffu);                   // global_addr[31:0]
  g0[3] = (unsigned)((ga >> 32) & 0x01ffffffu) | (2u << 30);  // addr[56:32] | type=2
  i32x8 g1;
  g1[0] = (int)((1u << 16) |                              // data_size = 2 bytes
                (1u << 20) |                              // pad_enable
                (pad_int << 22) | (pad_amt << 25));       // pad interval/amount codes
  g1[1] = (int)((tensor_d0 & 0xffffu) << 16);             // tensor_dim0[15:0]
  g1[2] = (int)(((tensor_d0 >> 16) & 0xffffu) |           // tensor_dim0[31:16]
                ((tensor_d1 & 0xffffu) << 16));           // tensor_dim1[15:0]
  g1[3] = (int)(((tensor_d1 >> 16) & 0xffffu) |           // tensor_dim1[31:16]
                ((tile0 & 0xffffu) << 16));               // tile_dim0
  g1[4] = (int)(tile1 & 0xffffu);                         // tile_dim1 (tile_dim2=0)
  g1[5] = (int)stride0;                                   // tensor_dim0_stride[31:0]
  g1[6] = 0;                                              // stride[47:32], dim1_stride lo
  g1[7] = 0;
  i32x4 z4 = (i32x4)0;
  i32x8 z8 = (i32x8)0;
  __builtin_amdgcn_tensor_load_to_lds(g0, g1, z4, z4, z8, 0);
}
#endif

// ---------------------------------------------------------------------------
// fp32 [R,C] -> bf16 [C,R]  (weights pre-transposed so WMMA B-frags are contiguous)
// ---------------------------------------------------------------------------
__global__ __launch_bounds__(256)
void transpose_to_bf16(const float* __restrict__ in, __bf16* __restrict__ out,
                       int R, int C) {
  __shared__ float tile[32][33];
  const int tx = threadIdx.x, ty = threadIdx.y;       // 32 x 8
  const int c0 = blockIdx.x * 32, r0 = blockIdx.y * 32;
#pragma unroll
  for (int i = 0; i < 4; ++i)
    tile[ty + i * 8][tx] = in[(size_t)(r0 + ty + i * 8) * C + c0 + tx];
  __syncthreads();
#pragma unroll
  for (int i = 0; i < 4; ++i)
    out[(size_t)(c0 + ty + i * 8) * R + r0 + tx] = (__bf16)tile[tx][ty + i * 8];
}

// ---------------------------------------------------------------------------
// Row LayerNorm: compile-time selected fp32 / bf16 outputs
// ---------------------------------------------------------------------------
template <bool OUTF, bool OUTB>
__global__ __launch_bounds__(256)
void ln_kernel(const float* __restrict__ x, const float* __restrict__ w,
               const float* __restrict__ b, float* __restrict__ yf,
               __bf16* __restrict__ yb, int E) {
  __shared__ float rs[256];
  __shared__ float rs2[256];
  const int row = blockIdx.x, t = threadIdx.x;
  const float* xr = x + (size_t)row * E;
  float s = 0.f, s2 = 0.f;
  for (int c = t; c < E; c += 256) { float v = xr[c]; s += v; s2 += v * v; }
  rs[t] = s; rs2[t] = s2;
  __syncthreads();
  for (int off = 128; off > 0; off >>= 1) {
    if (t < off) { rs[t] += rs[t + off]; rs2[t] += rs2[t + off]; }
    __syncthreads();
  }
  const float mu = rs[0] / E;
  const float var = rs2[0] / E - mu * mu;
  const float rstd = rsqrtf(var + 1e-5f);
  for (int c = t; c < E; c += 256) {
    float v = (xr[c] - mu) * rstd * w[c] + b[c];
    if (OUTF) yf[(size_t)row * E + c] = v;
    if (OUTB) yb[(size_t)row * E + c] = (__bf16)v;
  }
}

// ---------------------------------------------------------------------------
// bf16 WMMA GEMM: C[M,N] = A[M,K] @ BT[N,K]^T  (+bias, +gelu, +addend; all
// compile-time). 128x128 block tile, 8 waves, wave = 32x64 (2x4 of 16x16).
// TDM double-buffered pipeline: DMA of chunk i+1 overlaps WMMA on chunk i;
// s_wait_tensorcnt(2) retires chunk i while chunk i+1 stays in flight.
// ---------------------------------------------------------------------------
#define GS 72  // padded LDS K-stride (bf16 units): 144B = 64 elems + 16B TDM pad
template <bool HAS_BIAS, bool GELU_ACT, bool HAS_ADD, bool OUT_F, bool OUT_B>
__global__ __launch_bounds__(256)
void gemm_bf16_kernel(const __bf16* __restrict__ A, const __bf16* __restrict__ BT,
                      float* __restrict__ Cf, __bf16* __restrict__ Cb,
                      const float* __restrict__ bias, const float* __restrict__ addend,
                      int M, int N, int K) {
  __shared__ __bf16 As[2][128 * GS];
  __shared__ __bf16 Bs[2][128 * GS];
  const int t = threadIdx.x;
  const int wid = t >> 5, lane = t & 31, lm = lane & 15, lh = lane >> 4;
  const int m0 = blockIdx.y * 128, n0 = blockIdx.x * 128;
  const int wm = wid & 3, wn = wid >> 2;      // wave tile: rows 32*wm, cols 64*wn
  v8f acc[2][4];
#pragma unroll
  for (int mi = 0; mi < 2; ++mi)
#pragma unroll
    for (int ni = 0; ni < 4; ++ni)
#pragma unroll
      for (int r = 0; r < 8; ++r) acc[mi][ni][r] = 0.0f;

  const int nk = K >> 6;                      // 64-wide K chunks
#if HAVE_TDM
  if (wid == 0) {                             // prologue: DMA chunk 0 into buf 0
    tdm_load_2d((unsigned)(size_t)(void*)&As[0][0], &A[(size_t)m0 * K],
                64u, 128u, (unsigned)K, 4u, 3u);
    tdm_load_2d((unsigned)(size_t)(void*)&Bs[0][0], &BT[(size_t)n0 * K],
                64u, 128u, (unsigned)K, 4u, 3u);
  }
#endif
  for (int ic = 0; ic < nk; ++ic) {
    const int cur = ic & 1;
#if HAVE_TDM
    if (wid == 0) {
      if (ic + 1 < nk) {                      // prefetch chunk ic+1 into other buf
        const int nxt = cur ^ 1;
        tdm_load_2d((unsigned)(size_t)(void*)&As[nxt][0],
                    &A[(size_t)m0 * K + (ic + 1) * 64], 64u, 128u, (unsigned)K, 4u, 3u);
        tdm_load_2d((unsigned)(size_t)(void*)&Bs[nxt][0],
                    &BT[(size_t)n0 * K + (ic + 1) * 64], 64u, 128u, (unsigned)K, 4u, 3u);
        __builtin_amdgcn_s_wait_tensorcnt((short)2);   // chunk ic landed; ic+1 in flight
      } else {
        __builtin_amdgcn_s_wait_tensorcnt((short)0);
      }
    }
    __syncthreads();                          // chunk ic visible block-wide
#else
    __syncthreads();
    {
      const int kc = ic * 64;
#pragma unroll
      for (int i = 0; i < 4; ++i) {           // 1024 x 16B: A and B tiles
        int idx = t + i * 256;
        int row = idx >> 3, c8 = idx & 7;
        *(u32x4*)&As[cur][row * GS + c8 * 8] =
            *(const u32x4*)&A[(size_t)(m0 + row) * K + kc + c8 * 8];
        *(u32x4*)&Bs[cur][row * GS + c8 * 8] =
            *(const u32x4*)&BT[(size_t)(n0 + row) * K + kc + c8 * 8];
      }
    }
    __syncthreads();
#endif
#pragma unroll
    for (int kk = 0; kk < 2; ++kk) {
      v16bf af[2];
#pragma unroll
      for (int mi = 0; mi < 2; ++mi)
        af[mi] = ld_frag_a(&As[cur][(wm * 32 + mi * 16 + lm) * GS + kk * 32 + lh * 8]);
#pragma unroll
      for (int ni = 0; ni < 4; ++ni) {
        v16bf bf = ld_frag_b(&Bs[cur][(wn * 64 + ni * 16 + lm) * GS + kk * 32 + lh * 8]);
#pragma unroll
        for (int mi = 0; mi < 2; ++mi)
          acc[mi][ni] = wmma_bf16(af[mi], bf, acc[mi][ni]);
      }
    }
    __syncthreads();                          // all reads of buf cur done before reuse
  }
  // fused epilogue (fully compile-time configured: no per-element branches)
#pragma unroll
  for (int mi = 0; mi < 2; ++mi) {
#pragma unroll
    for (int ni = 0; ni < 4; ++ni) {
      const int col = n0 + wn * 64 + ni * 16 + lm;
      float bv = HAS_BIAS ? bias[col] : 0.0f;
#pragma unroll
      for (int r = 0; r < 8; ++r) {
        const int row = m0 + wm * 32 + mi * 16 + r + 8 * lh;
        float v = acc[mi][ni][r] + bv;
        if (GELU_ACT) v = 0.5f * v * (1.0f + erff(v * 0.70710678118654752f));
        if (HAS_ADD) v += addend[(size_t)row * N + col];
        if (OUT_F) Cf[(size_t)row * N + col] = v;
        if (OUT_B) Cb[(size_t)row * N + col] = (__bf16)v;
      }
    }
  }
}

// ---------------------------------------------------------------------------
// Flash attention, causal. Block = (b, head, 64-query tile), 4 waves x 16 rows.
// scores = (Q Kt)/sqrt(D) masked; online softmax; ctx = P V; all via WMMA.
// K tile staged by TDM (overlapped with the VALU V-transpose staging).
// ---------------------------------------------------------------------------
#define KTS 136  // K-tile LDS row stride (bf16): 272B = 128 elems + 16B TDM pad
#define VTS 72   // transposed-V LDS row stride
#define PS  72   // P staging stride
__global__ __launch_bounds__(128)
void attn_kernel(const __bf16* __restrict__ qb, const __bf16* __restrict__ kb,
                 const __bf16* __restrict__ vb, __bf16* __restrict__ ctxb) {
  __shared__ __bf16 Kt[64 * KTS];       // 64 keys x 128 d, row-major
  __shared__ __bf16 Vt[128 * VTS];      // 128 d x 64 keys (transposed)
  __shared__ __bf16 Pl[4 * 16 * PS];    // per-wave P staging (16 x 64)
  const int t = threadIdx.x;
  const int wid = t >> 5, lane = t & 31, lm = lane & 15, lh = lane >> 4;
  const int qblk = blockIdx.x, head = blockIdx.y, batch = blockIdx.z;
  const size_t rowQ0 = (size_t)batch * SS + qblk * 64;
  const float scale = 0.08838834764831845f;  // 1/sqrt(128)

  // Q fragments for this wave's 16 rows (kept in VGPRs for the whole kernel)
  v16bf qf[4];
  {
    const __bf16* qp = qb + (rowQ0 + wid * 16 + lm) * EE + head * DD;
#pragma unroll
    for (int kk = 0; kk < 4; ++kk) qf[kk] = ld_frag_a(qp + kk * 32 + lh * 8);
  }

  float m_r[8], l_r[8];
  v8f ctx[8];
#pragma unroll
  for (int r = 0; r < 8; ++r) { m_r[r] = -INFINITY; l_r[r] = 0.0f; }
#pragma unroll
  for (int nt = 0; nt < 8; ++nt)
#pragma unroll
    for (int r = 0; r < 8; ++r) ctx[nt][r] = 0.0f;

  const int qw = qblk * 64 + wid * 16;   // seq-local first q row of wave
  __bf16* Pw = &Pl[wid * 16 * PS];

  for (int j = 0; j <= qblk; ++j) {
    const int kblock = j * 64;
    const size_t rowK0 = (size_t)batch * SS + kblock;
    __syncthreads();
#if HAVE_TDM
    if (wid == 0)   // 64 rows x 128 bf16, pad codes: 64DW interval, 4DW pad
      tdm_load_2d((unsigned)(size_t)(void*)Kt,
                  &kb[rowK0 * EE + head * DD], 128u, 64u, (unsigned)EE, 5u, 3u);
#else
#pragma unroll
    for (int i = 0; i < 8; ++i) {
      int idx = t + i * 128;
      int row = idx >> 4, c16 = idx & 15;
      *(u32x4*)&Kt[row * KTS + c16 * 8] =
          *(const u32x4*)&kb[(rowK0 + row) * EE + head * DD + c16 * 8];
    }
#endif
    // stage V transposed -- B-frags for P@V read contiguous keys per lane
#pragma unroll
    for (int i = 0; i < 32; ++i) {
      int u = t + i * 128;
      int row = u >> 6, du = u & 63;
      union { unsigned int w; __bf16 h[2]; } cv;
      cv.w = *(const unsigned int*)&vb[(rowK0 + row) * EE + head * DD + du * 2];
      Vt[(2 * du) * VTS + row]     = cv.h[0];
      Vt[(2 * du + 1) * VTS + row] = cv.h[1];
    }
#if HAVE_TDM
    if (wid == 0) __builtin_amdgcn_s_wait_tensorcnt((short)0);
#endif
    __syncthreads();

    // scores: 16x64 per wave = 4 tiles x 4 d-chunks of WMMA
    v8f st[4];
#pragma unroll
    for (int nt = 0; nt < 4; ++nt) {
      v8f a;
#pragma unroll
      for (int r = 0; r < 8; ++r) a[r] = 0.0f;
#pragma unroll
      for (int kk = 0; kk < 4; ++kk) {
        v16bf bf = ld_frag_b(&Kt[(nt * 16 + lm) * KTS + kk * 32 + lh * 8]);
        a = wmma_bf16(qf[kk], bf, a);
      }
      st[nt] = a;
    }
    // scale then causal mask (== reference: mask -inf then /sqrt(D))
#pragma unroll
    for (int nt = 0; nt < 4; ++nt) {
      const int key = kblock + nt * 16 + lm;
#pragma unroll
      for (int r = 0; r < 8; ++r) {
        float s = st[nt][r] * scale;
        st[nt][r] = (key > qw + r + 8 * lh) ? -INFINITY : s;
      }
    }
    // online softmax; row m=r+8*lh lives in one 16-lane half -> xor-reduce
#pragma unroll
    for (int r = 0; r < 8; ++r) {
      float mx = st[0][r];
#pragma unroll
      for (int nt = 1; nt < 4; ++nt) mx = fmaxf(mx, st[nt][r]);
#pragma unroll
      for (int off = 1; off < 16; off <<= 1) mx = fmaxf(mx, __shfl_xor(mx, off, 32));
      const float mnew = fmaxf(m_r[r], mx);
      const float alpha = __expf(m_r[r] - mnew);
      float ps = 0.0f;
#pragma unroll
      for (int nt = 0; nt < 4; ++nt) {
        float pv = __expf(st[nt][r] - mnew);
        st[nt][r] = pv;
        ps += pv;
      }
#pragma unroll
      for (int off = 1; off < 16; off <<= 1) ps += __shfl_xor(ps, off, 32);
      l_r[r] = l_r[r] * alpha + ps;
      m_r[r] = mnew;
#pragma unroll
      for (int nt = 0; nt < 8; ++nt) ctx[nt][r] *= alpha;
    }
    // restripe P (C-layout) -> A-fragment layout via wave-private LDS
#pragma unroll
    for (int nt = 0; nt < 4; ++nt)
#pragma unroll
      for (int r = 0; r < 8; ++r)
        Pw[(r + 8 * lh) * PS + nt * 16 + lm] = (__bf16)st[nt][r];
    // ctx += P @ V : 2 k-chunks x 8 d-tiles
#pragma unroll
    for (int kk2 = 0; kk2 < 2; ++kk2) {
      v16bf pa = ld_frag_a(&Pw[lm * PS + kk2 * 32 + lh * 8]);
#pragma unroll
      for (int nt = 0; nt < 8; ++nt) {
        v16bf bf = ld_frag_b(&Vt[(nt * 16 + lm) * VTS + kk2 * 32 + lh * 8]);
        ctx[nt] = wmma_bf16(pa, bf, ctx[nt]);
      }
    }
  }
  // normalize + store ctx as bf16 [B*S, E] for the Wo GEMM
#pragma unroll
  for (int nt = 0; nt < 8; ++nt)
#pragma unroll
    for (int r = 0; r < 8; ++r) {
      const size_t row = rowQ0 + wid * 16 + r + 8 * lh;
      ctxb[row * EE + head * DD + nt * 16 + lm] = (__bf16)(ctx[nt][r] / l_r[r]);
    }
}

// ---------------------------------------------------------------------------
extern "C" void kernel_launch(void* const* d_in, const int* in_sizes, int n_in,
                              void* d_out, int out_size, void* d_ws, size_t ws_size,
                              hipStream_t stream) {
  (void)in_sizes; (void)n_in; (void)out_size; (void)ws_size;
  const float* x    = (const float*)d_in[0];
  const float* ln1w = (const float*)d_in[1];
  const float* ln1b = (const float*)d_in[2];
  const float* wq   = (const float*)d_in[3];
  const float* wk   = (const float*)d_in[4];
  const float* wv   = (const float*)d_in[5];
  const float* wo   = (const float*)d_in[6];
  const float* bo   = (const float*)d_in[7];
  const float* ln2w = (const float*)d_in[8];
  const float* ln2b = (const float*)d_in[9];
  const float* w1   = (const float*)d_in[10];
  const float* b1   = (const float*)d_in[11];
  const float* w2   = (const float*)d_in[12];
  const float* b2   = (const float*)d_in[13];
  float* out = (float*)d_out;   // also used as scratch for a = attn_out + x

  char* p = (char*)d_ws;
  float*  x1f  = (float*)p;  p += (size_t)NROWS * EE * 4;   // LN1 out fp32 (final residual)
  __bf16* x1b  = (__bf16*)p; p += (size_t)NROWS * EE * 2;
  __bf16* wqT  = (__bf16*)p; p += (size_t)EE * EE * 2;      // weights as bf16 [N,K]
  __bf16* wkT  = (__bf16*)p; p += (size_t)EE * EE * 2;
  __bf16* wvT  = (__bf16*)p; p += (size_t)EE * EE * 2;
  __bf16* woT  = (__bf16*)p; p += (size_t)EE * EE * 2;
  __bf16* w1T  = (__bf16*)p; p += (size_t)EE * FF * 2;
  __bf16* w2T  = (__bf16*)p; p += (size_t)EE * FF * 2;
  __bf16* qbuf = (__bf16*)p; p += (size_t)NROWS * EE * 2;
  __bf16* kbuf = (__bf16*)p; p += (size_t)NROWS * EE * 2;
  __bf16* vbuf = (__bf16*)p; p += (size_t)NROWS * EE * 2;
  __bf16* ctxb = (__bf16*)p; p += (size_t)NROWS * EE * 2;
  __bf16* ybuf = (__bf16*)p; p += (size_t)NROWS * EE * 2;   // LN2 out bf16
  __bf16* hbuf = (__bf16*)p; p += (size_t)NROWS * FF * 2;   // GELU(FFN1) bf16

  const dim3 tb(32, 8);
  transpose_to_bf16<<<dim3(EE / 32, EE / 32), tb, 0, stream>>>(wq, wqT, EE, EE);
  transpose_to_bf16<<<dim3(EE / 32, EE / 32), tb, 0, stream>>>(wk, wkT, EE, EE);
  transpose_to_bf16<<<dim3(EE / 32, EE / 32), tb, 0, stream>>>(wv, wvT, EE, EE);
  transpose_to_bf16<<<dim3(EE / 32, EE / 32), tb, 0, stream>>>(wo, woT, EE, EE);
  transpose_to_bf16<<<dim3(FF / 32, EE / 32), tb, 0, stream>>>(w1, w1T, EE, FF);
  transpose_to_bf16<<<dim3(EE / 32, FF / 32), tb, 0, stream>>>(w2, w2T, FF, EE);

  // x1 = LN1(x)
  ln_kernel<true, true><<<NROWS, 256, 0, stream>>>(x, ln1w, ln1b, x1f, x1b, EE);

  // Q, K, V projections (bf16 out only)
  gemm_bf16_kernel<false, false, false, false, true>
      <<<dim3(EE / 128, NROWS / 128), 256, 0, stream>>>(
      x1b, wqT, nullptr, qbuf, nullptr, nullptr, NROWS, EE, EE);
  gemm_bf16_kernel<false, false, false, false, true>
      <<<dim3(EE / 128, NROWS / 128), 256, 0, stream>>>(
      x1b, wkT, nullptr, kbuf, nullptr, nullptr, NROWS, EE, EE);
  gemm_bf16_kernel<false, false, false, false, true>
      <<<dim3(EE / 128, NROWS / 128), 256, 0, stream>>>(
      x1b, wvT, nullptr, vbuf, nullptr, nullptr, NROWS, EE, EE);

  // causal flash attention
  attn_kernel<<<dim3(SS / 64, HH, BB), 128, 0, stream>>>(qbuf, kbuf, vbuf, ctxb);

  // a = ctx @ wo + bo + x   (stored in d_out, consumed by LN2)
  gemm_bf16_kernel<true, false, true, true, false>
      <<<dim3(EE / 128, NROWS / 128), 256, 0, stream>>>(
      ctxb, woT, out, nullptr, bo, x, NROWS, EE, EE);

  // y = LN2(a)
  ln_kernel<false, true><<<NROWS, 256, 0, stream>>>(out, ln2w, ln2b, nullptr, ybuf, EE);

  // h = gelu(y @ w1 + b1)
  gemm_bf16_kernel<true, true, false, false, true>
      <<<dim3(FF / 128, NROWS / 128), 256, 0, stream>>>(
      ybuf, w1T, nullptr, hbuf, b1, nullptr, NROWS, FF, EE);

  // out = x1 + (h @ w2 + b2)
  gemm_bf16_kernel<true, false, true, true, false>
      <<<dim3(EE / 128, NROWS / 128), 256, 0, stream>>>(
      hbuf, w2T, out, nullptr, b2, x1f, NROWS, EE, FF);
}
